// MixHopNet_84507776516698
// MI455X (gfx1250) — compile-verified, matrix operand order checked by reading, not run on hardware
//
#include <hip/hip_runtime.h>
#include <math.h>

typedef __bf16 bf16_t;
typedef __attribute__((ext_vector_type(16))) __bf16 v16bf;
typedef __attribute__((ext_vector_type(8)))  float  v8f;

#define F_IN  128
#define HID   512
#define NCLS  40
#define L1OUT 1536   // 3*HID
#define L2OUT 80     // 2*NCLS

#ifndef __has_builtin
#define __has_builtin(x) 0
#endif
#if __has_builtin(__builtin_amdgcn_global_load_async_to_lds_b128)
#define USE_ASYNC_LDS 1
// exact parameter types per hipcc diagnostic: v4i (gcc vector_size) pointers,
// global source in AS(1), LDS destination in AS(3)
typedef int v4i_vs __attribute__((vector_size(16)));
typedef __attribute__((address_space(1))) v4i_vs* gptr_v4i;
typedef __attribute__((address_space(3))) v4i_vs* lptr_v4i;
#else
#define USE_ASYNC_LDS 0
#endif
#if __has_builtin(__builtin_amdgcn_s_wait_asynccnt)
#define WAIT_ASYNC() __builtin_amdgcn_s_wait_asynccnt(0)
#else
#define WAIT_ASYNC() asm volatile("s_wait_asynccnt 0" ::: "memory")
#endif

static __device__ __forceinline__ int imin(int a, int b) { return a < b ? a : b; }

// ---------------- zero / degree / normalization ----------------
__global__ void k_zero(int* p, int n) {
  int i = blockIdx.x * 256 + threadIdx.x;
  if (i < n) p[i] = 0;
}
__global__ void k_count(int* __restrict__ cnt, const int* __restrict__ col, int E) {
  int e = blockIdx.x * 256 + threadIdx.x;
  if (e < E) atomicAdd(&cnt[col[e]], 1);
}
__global__ void k_dinv(float* __restrict__ dinv, const int* __restrict__ cnt, int n) {
  int i = blockIdx.x * 256 + threadIdx.x;
  if (i < n) dinv[i] = rsqrtf((float)cnt[i] + 1.0f);   // +1 = self loop
}

// ---------------- prefix scan: cnt -> rowptr (exclusive) ----------------
__global__ void k_scan1(const int* __restrict__ cnt, int* __restrict__ rowptr,
                        int* __restrict__ bsum, int n) {
  __shared__ int s[256];
  int tid = threadIdx.x;
  int i = blockIdx.x * 256 + tid;
  int v = (i < n) ? cnt[i] : 0;
  s[tid] = v;
  __syncthreads();
  for (int off = 1; off < 256; off <<= 1) {
    int t = (tid >= off) ? s[tid - off] : 0;
    __syncthreads();
    s[tid] += t;
    __syncthreads();
  }
  if (i < n) rowptr[i] = s[tid] - v;               // exclusive within block
  if (tid == 255) bsum[blockIdx.x] = s[255];
}
__global__ void k_scan2(int* bsum, int nb) {       // single block of 1024
  __shared__ int s[1024];
  int tid = threadIdx.x;
  int v = (tid < nb) ? bsum[tid] : 0;
  s[tid] = v;
  __syncthreads();
  for (int off = 1; off < 1024; off <<= 1) {
    int t = (tid >= off) ? s[tid - off] : 0;
    __syncthreads();
    s[tid] += t;
    __syncthreads();
  }
  if (tid < nb) bsum[tid] = s[tid] - v;            // exclusive block offsets
}
__global__ void k_scan3(int* __restrict__ rowptr, const int* __restrict__ bsum,
                        int n, int E) {
  int i = blockIdx.x * 256 + threadIdx.x;
  if (i < n) rowptr[i] += bsum[i >> 8];
  if (i == n) rowptr[n] = E;
}

// ---------------- CSR fill (dest-sorted edge list with weights) ----------------
__global__ void k_fill(const int* __restrict__ row, const int* __restrict__ col,
                       const float* __restrict__ dinv,
                       const int* __restrict__ rowptr, int* __restrict__ cursor,
                       int* __restrict__ esrc, float* __restrict__ ew, int E) {
  int e = blockIdx.x * 256 + threadIdx.x;
  if (e >= E) return;
  int r = row[e], c = col[e];
  int slot = rowptr[c] + atomicAdd(&cursor[c], 1);
  esrc[slot] = r;
  ew[slot] = dinv[r] * dinv[c];
}

// ---------------- gather SpMM, 128 features (wave per node, float4 lanes) ----------
__global__ void k_spmm128g(float* __restrict__ y, const float* __restrict__ x,
                           const int* __restrict__ rowptr, const int* __restrict__ esrc,
                           const float* __restrict__ ew, const float* __restrict__ dinv,
                           int n) {
  int i = blockIdx.x * 8 + (threadIdx.x >> 5);
  int lane = threadIdx.x & 31;
  if (i >= n) return;
  float d = dinv[i], w0 = d * d;
  float4 v = ((const float4*)(x + (long)i * F_IN))[lane];
  float4 acc = make_float4(w0 * v.x, w0 * v.y, w0 * v.z, w0 * v.w);  // self loop
  int e0 = rowptr[i], e1 = rowptr[i + 1];
  for (int j = e0; j < e1; ++j) {
    int s = esrc[j];
    float w = ew[j];
    if (j + 1 < e1)
      __builtin_prefetch((const void*)(x + (long)esrc[j + 1] * F_IN), 0, 1);
    float4 u = ((const float4*)(x + (long)s * F_IN))[lane];
    acc.x += w * u.x; acc.y += w * u.y; acc.z += w * u.z; acc.w += w * u.w;
  }
  ((float4*)(y + (long)i * F_IN))[lane] = acc;
}

// ---------------- gather SpMM, 40 features + bias, writes out[:,40:80] ----------
__global__ void k_spmm40g(float* __restrict__ out, const float* __restrict__ u1,
                          const int* __restrict__ rowptr, const int* __restrict__ esrc,
                          const float* __restrict__ ew, const float* __restrict__ dinv,
                          const float* __restrict__ b2hi, int n) {
  int i = blockIdx.x * 8 + (threadIdx.x >> 5);
  int lane = threadIdx.x & 31;
  if (i >= n) return;
  float d = dinv[i], w0 = d * d;
  const float* ui = u1 + (long)i * NCLS;
  float a0 = w0 * ui[lane];                                  // f = lane (0..31)
  float a1 = (lane < 8) ? w0 * ui[32 + lane] : 0.0f;         // f = 32+lane
  int e0 = rowptr[i], e1 = rowptr[i + 1];
  for (int j = e0; j < e1; ++j) {
    int s = esrc[j];
    float w = ew[j];
    const float* us = u1 + (long)s * NCLS;
    a0 += w * us[lane];
    if (lane < 8) a1 += w * us[32 + lane];
  }
  float* o = out + (long)i * L2OUT + NCLS;
  o[lane] = a0 + b2hi[lane];
  if (lane < 8) o[32 + lane] = a1 + b2hi[32 + lane];
}

// ---------------- weight pre-pack: f32 KxN -> bf16 WMMA-fragment layout ----------
// Packed index: (((ct * (K/32) + ks) * 32 + lane) * 16 + j)
// j in 0..7  -> K = ks*32 + half*8 + j        (half = lane>=16)
// j in 8..15 -> K = ks*32 + 16 + half*8 + (j-8),  N = ct*16 + lane%16
__global__ void k_packB(const float* __restrict__ B, bf16_t* __restrict__ Bp,
                        int K, int Nout, int ctiles) {
  long idx = (long)blockIdx.x * 256 + threadIdx.x;
  long total = (long)ctiles * (K >> 5) * 512;
  if (idx >= total) return;
  int j    = (int)(idx & 15);
  int lane = (int)((idx >> 4) & 31);
  long rest = idx >> 9;
  int ks = (int)(rest % (K >> 5));
  int ct = (int)(rest / (K >> 5));
  int half = lane >> 4, l16 = lane & 15;
  int k = ks * 32 + (j >> 3) * 16 + half * 8 + (j & 7);
  int nn = ct * 16 + l16;
  if (nn >= Nout) nn = Nout - 1;
  Bp[idx] = (bf16_t)B[(long)k * Nout + nn];
}

// ---------------- WMMA bf16 GEMM: C = A(MxK) @ B(KxN) (+bias, relu) ----------------
// 256 threads = 8 waves; one 16-row M-tile per block; wave w owns col tile
// blockIdx.y*8 + w. A-panel staged through LDS as bf16 in K-chunks of 128;
// bf16 A uses GLOBAL_LOAD_ASYNC_TO_LDS_B128 (ASYNCcnt-tracked) when available.
template <typename AT, typename OT, bool RELU, bool HASBIAS>
__global__ __launch_bounds__(256)
void k_gemm_wmma(const AT* __restrict__ A, int lda,
                 const bf16_t* __restrict__ Bp,      // packed fragments
                 OT* __restrict__ C, int ldc,
                 const float* __restrict__ bias,
                 int M, int K, int Nout) {
  __shared__ __align__(32) bf16_t As[16 * 128];
  const int wave = threadIdx.x >> 5;
  const int lane = threadIdx.x & 31;
  const int half = lane >> 4;
  const int l16  = lane & 15;
  const int rowbase = blockIdx.x * 16;
  const int coltile = blockIdx.y * 8 + wave;
  const int ctiles  = (Nout + 15) >> 4;
  const int ct = imin(coltile, ctiles - 1);
  const bf16_t* Bct = Bp + (long)ct * (K >> 5) * 512;

  // cooperative A staging coords: each thread handles 8 contiguous elements (16B bf16)
  const int rr = threadIdx.x >> 4;          // 0..15
  const int cc = (threadIdx.x & 15) * 8;    // 0..120
  const int rg = imin(rowbase + rr, M - 1);

  v8f acc = {};
  for (int k0 = 0; k0 < K; k0 += 128) {
    const AT* ap = A + (long)rg * lda + k0 + cc;
    bf16_t* dst = &As[rr * 128 + cc];
    if constexpr (sizeof(AT) == 2) {
#if USE_ASYNC_LDS
      __builtin_amdgcn_global_load_async_to_lds_b128(
          (gptr_v4i)ap, (lptr_v4i)dst, 0, 0);
      WAIT_ASYNC();
#else
      #pragma unroll
      for (int i = 0; i < 8; ++i) dst[i] = (bf16_t)ap[i];
#endif
    } else {
      #pragma unroll
      for (int i = 0; i < 8; ++i) dst[i] = (bf16_t)ap[i];
    }
    __syncthreads();
    #pragma unroll
    for (int kk = 0; kk < 128; kk += 32) {
      v16bf a, b;
      const bf16_t* af = &As[l16 * 128 + kk + half * 8];
      #pragma unroll
      for (int i = 0; i < 8; ++i) { a[i] = af[i]; a[8 + i] = af[16 + i]; }
      int ksg = (k0 + kk) >> 5;
      b = *(const v16bf*)(Bct + ((long)ksg * 32 + lane) * 16);
      acc = __builtin_amdgcn_wmma_f32_16x16x32_bf16(
          false, a, false, b, (short)0, acc, false, false);
    }
    __syncthreads();
  }
  #pragma unroll
  for (int v = 0; v < 8; ++v) {
    int r = rowbase + v + 8 * half;
    int c = coltile * 16 + l16;
    if (r < M && c < Nout) {
      float val = acc[v];
      if constexpr (HASBIAS) val += bias[c];
      if constexpr (RELU) val = fmaxf(val, 0.0f);
      C[(long)r * ldc + c] = (OT)val;
    }
  }
}

// ---------------- log_softmax over 80 cols, one wave32 per row ----------------
__global__ void k_logsoftmax80(float* __restrict__ out, int n) {
  int r = blockIdx.x * 8 + (threadIdx.x >> 5);
  int lane = threadIdx.x & 31;
  if (r >= n) return;
  float* o = out + (long)r * L2OUT;
  float v0 = o[lane], v1 = o[lane + 32];
  float v2 = (lane < 16) ? o[lane + 64] : -__builtin_inff();
  float m = fmaxf(fmaxf(v0, v1), v2);
  #pragma unroll
  for (int off = 16; off; off >>= 1) m = fmaxf(m, __shfl_xor(m, off, 32));
  float s = __expf(v0 - m) + __expf(v1 - m) + ((lane < 16) ? __expf(v2 - m) : 0.0f);
  #pragma unroll
  for (int off = 16; off; off >>= 1) s += __shfl_xor(s, off, 32);
  float lse = m + __logf(s);
  o[lane] = v0 - lse;
  o[lane + 32] = v1 - lse;
  if (lane < 16) o[lane + 64] = v2 - lse;
}

extern "C" void kernel_launch(void* const* d_in, const int* in_sizes, int n_in,
                              void* d_out, int out_size, void* d_ws, size_t ws_size,
                              hipStream_t stream) {
  const float* x  = (const float*)d_in[0];
  const int*   ei = (const int*)d_in[1];
  const float* w1 = (const float*)d_in[2];
  const float* b1 = (const float*)d_in[3];
  const float* w2 = (const float*)d_in[4];
  const float* b2 = (const float*)d_in[5];
  float* out = (float*)d_out;

  const int N = in_sizes[0] / F_IN;
  const int E = in_sizes[1] / 2;
  const int* row = ei;
  const int* col = ei + E;

  // ---- workspace carve-out (256B aligned regions) ----
  char* base = (char*)d_ws;
  size_t off = 0;
  auto alloc = [&](size_t bytes) -> void* {
    void* p = base + off;
    off += (bytes + 255) & ~(size_t)255;
    return p;
  };
  float*  dinv   = (float*)alloc((size_t)N * 4);
  float*  ax1    = (float*)alloc((size_t)N * F_IN * 4);
  float*  ax2    = (float*)alloc((size_t)N * F_IN * 4);
  float*  u1     = (float*)alloc((size_t)N * NCLS * 4);
  int*    rowptr = (int*)  alloc((size_t)(N + 1) * 4);
  int*    esrc   = (int*)  alloc((size_t)E * 4);
  float*  ew     = (float*)alloc((size_t)E * 4);
  int*    cnt    = (int*)  alloc((size_t)N * 4);
  int*    cursor = (int*)  alloc((size_t)N * 4);
  int*    bsum   = (int*)  alloc(1024 * 4);
  const long W1P = 32L * (F_IN / 32) * 512;            // 65536 per power
  const long W2P = 3L * (L1OUT / 32) * 512;            // 73728 per power
  bf16_t* w1p    = (bf16_t*)alloc((size_t)(3 * W1P) * 2);
  bf16_t* w2p    = (bf16_t*)alloc((size_t)(2 * W2P) * 2);
  bf16_t* h      = (bf16_t*)alloc((size_t)N * L1OUT * 2);

  const int nb = (N + 255) / 256;                      // scan blocks (<=1024)

  // ---- degree + dinv + CSR (dest-sorted) ----
  k_zero<<<nb, 256, 0, stream>>>(cnt, N);
  k_count<<<(E + 255) / 256, 256, 0, stream>>>(cnt, col, E);
  k_dinv<<<nb, 256, 0, stream>>>(dinv, cnt, N);
  k_scan1<<<nb, 256, 0, stream>>>(cnt, rowptr, bsum, N);
  k_scan2<<<1, 1024, 0, stream>>>(bsum, nb);
  k_scan3<<<(N + 256) / 256, 256, 0, stream>>>(rowptr, bsum, N, E);
  k_zero<<<nb, 256, 0, stream>>>(cursor, N);
  k_fill<<<(E + 255) / 256, 256, 0, stream>>>(row, col, dinv, rowptr, cursor,
                                              esrc, ew, E);

  // ---- pre-pack weights into WMMA fragment layout (bf16) ----
  for (int p = 0; p < 3; ++p)
    k_packB<<<(int)((W1P + 255) / 256), 256, 0, stream>>>(
        w1 + (long)p * F_IN * HID, w1p + p * W1P, F_IN, HID, HID / 16);
  for (int p = 0; p < 2; ++p)
    k_packB<<<(int)((W2P + 255) / 256), 256, 0, stream>>>(
        w2 + (long)p * L1OUT * NCLS, w2p + p * W2P, L1OUT, NCLS, 3);

  // ---- propagation: ax1 = A_hat x ; ax2 = A_hat ax1 (gather, no atomics) ----
  int gblocks = (N + 7) / 8;
  k_spmm128g<<<gblocks, 256, 0, stream>>>(ax1, x, rowptr, esrc, ew, dinv, N);
  k_spmm128g<<<gblocks, 256, 0, stream>>>(ax2, ax1, rowptr, esrc, ew, dinv, N);

  // ---- layer 1: h[:, p*512:(p+1)*512] = relu(src_p @ W1[p] + b1_p) (bf16) ----
  int mt = (N + 15) / 16;
  const float* srcs[3] = {x, ax1, ax2};
  for (int p = 0; p < 3; ++p) {
    k_gemm_wmma<float, bf16_t, true, true><<<dim3(mt, HID / 128), 256, 0, stream>>>(
        srcs[p], F_IN, w1p + p * W1P, h + (long)p * HID, L1OUT,
        b1 + (long)p * HID, N, F_IN, HID);
  }

  // ---- layer 2 GEMMs (reordered: GEMM before SpMM, 1536 -> 40 features) ----
  k_gemm_wmma<bf16_t, float, false, true><<<dim3(mt, 1), 256, 0, stream>>>(
      h, L1OUT, w2p, out, L2OUT, b2, N, L1OUT, NCLS);
  k_gemm_wmma<bf16_t, float, false, false><<<dim3(mt, 1), 256, 0, stream>>>(
      h, L1OUT, w2p + W2P, u1, NCLS, nullptr, N, L1OUT, NCLS);

  // ---- out[:, 40:80] = A_hat u1 + b2[40:80] (gather) ----
  k_spmm40g<<<gblocks, 256, 0, stream>>>(out, u1, rowptr, esrc, ew, dinv,
                                         b2 + NCLS, N);

  // ---- log_softmax over 80 classes ----
  k_logsoftmax80<<<(N + 7) / 8, 256, 0, stream>>>(out, N);
}
